// Transformer_75917841924387
// MI455X (gfx1250) — matrix operand order repaired and checked by reading, not compiled, not run on hardware
//
#include <hip/hip_runtime.h>
#include <hip/hip_bf16.h>
#include <math.h>

// ---------------------------------------------------------------------------
// Transformer encoder forward, MI455X (gfx1250, wave32, WMMA).
// All GEMMs: bf16 A/B, f32 accumulate via v_wmma_f32_16x16x32_bf16.
// GEMM: 64x256 block tile, 32x64 per-wave tile, A staged in LDS with
// double-buffered async global->LDS copies (ASYNCcnt) when available.
// Attention is fused (scores -> bias -> softmax -> P@V) in LDS per
// (batch, head, 16-query-rows) block: avoids ~1.2 GB of S/P global traffic.
// ---------------------------------------------------------------------------

typedef __bf16 bf16;
typedef __bf16 v16bf __attribute__((ext_vector_type(16)));
typedef __bf16 v8bf  __attribute__((ext_vector_type(8)));
typedef float  v8f   __attribute__((ext_vector_type(8)));
typedef int    v4i   __attribute__((ext_vector_type(4)));

#if defined(__AMDGCN__) && __has_builtin(__builtin_amdgcn_global_load_async_to_lds_b128) && __has_builtin(__builtin_amdgcn_s_wait_asynccnt)
#define USE_ASYNC 1
// builtin param 0 is v4i* in the global (device) address space; param 1 is the
// LDS destination.  C-style casts generic->AS1/AS3 lower to addrspacecast.
#define ASYNC_GLB(p) ((__attribute__((address_space(1))) v4i*)(p))
#define ASYNC_LDS(p) ((__attribute__((address_space(3))) v4i*)(p))
#else
#define USE_ASYNC 0
#endif

constexpr int kL = 6, kB = 16, kT = 512, kD = 512, kH = 8, kA = 64, kF = 2048;
constexpr int kBiasDim = 4;
constexpr int kHA = kH * kA;            // 512
constexpr int kRows = kB * kT;          // 8192
constexpr float kLnEps = 1e-3f;
constexpr float kNegInf = -3.402823466e38f;

// ---------------- WMMA fragment loaders ------------------------------------
// A-matrix 16x32 bf16 (ISA 7.12.2): lanes 0-15 row M=lane, elems 0..7 = K0..7,
// elems 8..15 = K16..23; lanes 16-31 row M=lane-16, K8..15 / K24..31.
__device__ __forceinline__ v16bf frag_a(const bf16* A, int lda, int row0, int k0, int lane) {
  const bf16* base = A + (size_t)(row0 + (lane & 15)) * lda + (k0 + ((lane >> 4) << 3));
  v8bf lo = *(const v8bf*)(base);
  v8bf hi = *(const v8bf*)(base + 16);
  return __builtin_shufflevector(lo, hi, 0,1,2,3,4,5,6,7,8,9,10,11,12,13,14,15);
}

// B-matrix 32x16 bf16, stored TRANSPOSED as Bt[N][K] so K is contiguous:
// lane n = lane&15 selects the column; lanes 0-15 hold K0..7 / K16..23, lanes
// 16-31 hold K8..15 / K24..31 (same order as frag_a) = two 16B loads.
__device__ __forceinline__ v16bf frag_bt(const bf16* Bt, int ldk, int k0, int col0, int lane) {
  const bf16* base = Bt + (size_t)(col0 + (lane & 15)) * ldk + (k0 + ((lane >> 4) << 3));
  v8bf lo = *(const v8bf*)(base);
  v8bf hi = *(const v8bf*)(base + 16);
  return __builtin_shufflevector(lo, hi, 0,1,2,3,4,5,6,7,8,9,10,11,12,13,14,15);
}

__device__ __forceinline__ v8f wmma_bf16(v16bf a, v16bf b, v8f c) {
  return __builtin_amdgcn_wmma_f32_16x16x32_bf16(false, a, false, b, (short)0, c, false, false);
}

// ---------------- elementwise kernels ---------------------------------------
__global__ __launch_bounds__(256) void fill_zero_kernel(float* p, int n) {
  int i = blockIdx.x * 256 + threadIdx.x;
  if (i < n) p[i] = 0.0f;
}

// x[b,t,d] = embed[token]*sqrt(D) + positional_encoding(t,d)
__global__ __launch_bounds__(256) void embed_kernel(const int* __restrict__ tokens,
                                                    const float* __restrict__ emb,
                                                    float* __restrict__ x) {
  int idx = blockIdx.x * 256 + threadIdx.x;
  if (idx >= kRows * kD) return;
  int d  = idx & (kD - 1);
  int bt = idx >> 9;
  int t  = bt & (kT - 1);
  int tok = tokens[bt];
  float di = (float)(2 * (d >> 1));
  float angle = (float)t * powf(10000.0f, -di / (float)kD);
  float pe = (d & 1) ? cosf(angle) : sinf(angle);
  x[idx] = emb[(size_t)tok * kD + d] * 22.62741699796952f + pe;  // sqrt(512)
}

// convert + transpose W[K][N] f32 -> Wt[N][K] bf16 (one time, uncoalesced OK)
__global__ __launch_bounds__(256) void convT_kernel(const float* __restrict__ W,
                                                    bf16* __restrict__ Wt, int K, int N) {
  size_t idx = (size_t)blockIdx.x * 256 + threadIdx.x;
  if (idx >= (size_t)K * N) return;
  int n = (int)(idx % N);
  int k = (int)(idx / N);
  Wt[(size_t)n * K + k] = (bf16)W[idx];
}

// LayerNorm: one wave per row of 512, 16 elems/lane, wave32 shuffle reduce.
template <bool BF16OUT>
__global__ __launch_bounds__(256) void ln_kernel(const float* __restrict__ x,
                                                 const float* __restrict__ g,
                                                 const float* __restrict__ bb,
                                                 float* __restrict__ outf,
                                                 bf16* __restrict__ outb) {
  int row  = blockIdx.x * 8 + (threadIdx.x >> 5);
  int lane = threadIdx.x & 31;
  const float* xr = x + (size_t)row * kD;
  float vals[16];
  float s = 0.0f;
#pragma unroll
  for (int i = 0; i < 16; ++i) { vals[i] = xr[lane + 32 * i]; s += vals[i]; }
#pragma unroll
  for (int o = 1; o < 32; o <<= 1) s += __shfl_xor(s, o, 32);
  float mu = s * (1.0f / kD);
  float v = 0.0f;
#pragma unroll
  for (int i = 0; i < 16; ++i) { float d = vals[i] - mu; v += d * d; }
#pragma unroll
  for (int o = 1; o < 32; o <<= 1) v += __shfl_xor(v, o, 32);
  float inv = rsqrtf(v * (1.0f / kD) + kLnEps);
#pragma unroll
  for (int i = 0; i < 16; ++i) {
    int c = lane + 32 * i;
    float r = (vals[i] - mu) * inv * g[c] + bb[c];
    if (BF16OUT) outb[(size_t)row * kD + c] = (bf16)r;
    else         outf[(size_t)row * kD + c] = r;
  }
}

// ksum[b,t,h] = sum_a k[b,t,h,a]
__global__ __launch_bounds__(256) void ksum_kernel(const bf16* __restrict__ kmat,
                                                   float* __restrict__ ksum) {
  int idx = blockIdx.x * 256 + threadIdx.x;
  if (idx >= kB * kT * kH) return;
  int h = idx & (kH - 1);
  int bt = idx >> 3;
  const bf16* p = kmat + (size_t)bt * kHA + h * kA;
  float s = 0.0f;
#pragma unroll
  for (int a = 0; a < kA; ++a) s += (float)p[a];
  ksum[idx] = s;
}

// vT[b,h,a,t] = v[b,t,h,a]  (so V can be the WMMA B-operand with contiguous K)
__global__ __launch_bounds__(256) void vtrans_kernel(const bf16* __restrict__ v,
                                                     bf16* __restrict__ vT) {
  int idx = blockIdx.x * 256 + threadIdx.x;
  if (idx >= kB * kT * kHA) return;
  int a = idx & (kA - 1);
  int h = (idx >> 6) & (kH - 1);
  int t = (idx >> 9) & (kT - 1);
  int b = idx >> 18;
  vT[(((size_t)b * kH + h) * kA + a) * kT + t] = v[idx];
}

// dense[b,q,k] += dot(be[type], bs) for each edge
__global__ __launch_bounds__(256) void bias_scatter_kernel(const int* __restrict__ ab,
                                                           const float* __restrict__ be,
                                                           const float* __restrict__ bs,
                                                           float* __restrict__ dense, int E) {
  int e = blockIdx.x * 256 + threadIdx.x;
  if (e >= E) return;
  int type = ab[e * 4 + 0], b = ab[e * 4 + 1], qi = ab[e * 4 + 2], ki = ab[e * 4 + 3];
  const float* bev = be + (size_t)type * kA;
  float ev = 0.0f;
#pragma unroll
  for (int a = 0; a < kA; ++a) ev += bev[a] * bs[a];
  atomicAdd(dense + ((size_t)b * kT + qi) * kT + ki, ev);
}

// ---------------- generic WMMA GEMM -----------------------------------------
// C[M,N] = A[M,K](bf16) @ Bt[N,K](bf16)  (+bias) with epilogue:
//   MODE 0: Cb = acc (bf16)                       (QKV projections)
//   MODE 1: Cf += acc + bias (residual add, f32)  (Wo, FFN2)
//   MODE 2: Cb = relu(acc + bias) (bf16)          (FFN1)
// Block = 8 waves (2x4); wave tile 32x64 (8 accumulators); block tile 64x256.
// A block-tile (64x32 bf16 = 4KB) is staged in LDS, double-buffered with
// async global->LDS copies (ASYNCcnt) when the builtin exists.
// grid = (N/256, M/64).
template <int MODE>
__global__ __launch_bounds__(256) void gemm_kernel(const bf16* __restrict__ A,
                                                   const bf16* __restrict__ Bt,
                                                   const float* __restrict__ bias,
                                                   float* __restrict__ Cf,
                                                   bf16* __restrict__ Cb,
                                                   int M, int N, int K) {
  __shared__ bf16 As[2][64][40];  // k-stride 40 keeps 16B alignment, dodges banks
  int tid = threadIdx.x, wave = tid >> 5, lane = tid & 31;
  int wm = wave >> 2, wn = wave & 3;
  int rowB = blockIdx.y * 64;
  int row0 = rowB + wm * 32;
  int col0 = blockIdx.x * 256 + wn * 64;

  // staging map: thread -> one 16B chunk of the 64x32 A tile
  int srow = tid >> 2, skoff = (tid & 3) * 8;
  const bf16* gsrc = A + (size_t)(rowB + srow) * K + skoff;

  v8f acc[2][4] = {};
  const int nsteps = K >> 5;

  // prime buffer 0
#if USE_ASYNC
  __builtin_amdgcn_global_load_async_to_lds_b128(
      ASYNC_GLB(gsrc), ASYNC_LDS(&As[0][srow][skoff]), 0, 0);
#else
  *(v8bf*)&As[0][srow][skoff] = *(const v8bf*)(gsrc);
#endif

  for (int s = 0; s < nsteps; ++s) {
    int buf = s & 1;
    int k0 = s << 5;
#if USE_ASYNC
    __builtin_amdgcn_s_wait_asynccnt(0);
#endif
    __syncthreads();   // buf is full; everyone done reading buf^1
    if (s + 1 < nsteps) {
#if USE_ASYNC
      __builtin_amdgcn_global_load_async_to_lds_b128(
          ASYNC_GLB(gsrc + k0 + 32), ASYNC_LDS(&As[buf ^ 1][srow][skoff]), 0, 0);
#else
      *(v8bf*)&As[buf ^ 1][srow][skoff] = *(const v8bf*)(gsrc + k0 + 32);
#endif
    }
    v16bf a0 = frag_a(&As[buf][0][0], 40, wm * 32, 0, lane);
    v16bf a1 = frag_a(&As[buf][0][0], 40, wm * 32 + 16, 0, lane);
#pragma unroll
    for (int j = 0; j < 4; ++j) {
      const bf16* bcol = Bt + (size_t)(col0 + j * 16 + (lane & 15)) * K;
      if (s + 1 < nsteps) __builtin_prefetch(bcol + k0 + 32, 0, 3);
      v16bf bfj = frag_bt(Bt, K, k0, col0 + j * 16, lane);
      acc[0][j] = wmma_bf16(a0, bfj, acc[0][j]);
      acc[1][j] = wmma_bf16(a1, bfj, acc[1][j]);
    }
  }

  // epilogue
#pragma unroll
  for (int sub = 0; sub < 2; ++sub) {
    int rb = row0 + sub * 16 + ((lane >> 4) << 3);
#pragma unroll
    for (int j = 0; j < 4; ++j) {
      int c = col0 + j * 16 + (lane & 15);
      float bv = (MODE != 0 && bias != nullptr) ? bias[c] : 0.0f;
#pragma unroll
      for (int i = 0; i < 8; ++i) {
        size_t off = (size_t)(rb + i) * N + c;
        float r = acc[sub][j][i] + bv;
        if (MODE == 0)      Cb[off] = (bf16)r;
        else if (MODE == 1) Cf[off] += r;
        else                Cb[off] = (bf16)fmaxf(r, 0.0f);
      }
    }
  }
}

// ---------------- fused attention -------------------------------------------
// One block = (b, h, 16 query rows).  Phase1: S[16,512] via WMMA into LDS.
// Phase2: edge-bias + scale + mask + softmax in LDS.  Phase3: P@V via WMMA
// with P read straight from LDS.  No global S/P traffic.
__global__ __launch_bounds__(256) void attn_kernel(const bf16* __restrict__ q,
                                                   const bf16* __restrict__ kmat,
                                                   const bf16* __restrict__ vT,
                                                   const float* __restrict__ dense,
                                                   const float* __restrict__ ksum,
                                                   const float* __restrict__ masks,
                                                   bf16* __restrict__ ctx) {
  __shared__ float S[16][516];
  __shared__ bf16  P[16][520];
  int bid = blockIdx.x;
  int qt = bid & 31;                 // T/16 = 32 query tiles
  int hh = (bid >> 5) & (kH - 1);
  int b  = bid >> 8;
  int tid = threadIdx.x, lane = tid & 31, wave = tid >> 5;
  int q0 = qt * 16;
  const bf16* Qb = q    + (size_t)b * kT * kHA + hh * kA;  // row stride kHA
  const bf16* Kb = kmat + (size_t)b * kT * kHA + hh * kA;

  // Phase 1: S = Q Kt  (8 waves x 4 column tiles each)
  for (int nt = wave; nt < 32; nt += 8) {
    v8f acc = v8f{0};
    for (int k0 = 0; k0 < kA; k0 += 32) {
      v16bf af  = frag_a(Qb, kHA, q0, k0, lane);
      v16bf bfj = frag_bt(Kb, kHA, k0, nt * 16, lane);  // "rows" of Kb are key cols
      acc = wmma_bf16(af, bfj, acc);
    }
    int col = nt * 16 + (lane & 15);
    int rb = (lane >> 4) << 3;
#pragma unroll
    for (int i = 0; i < 8; ++i) S[rb + i][col] = acc[i];
  }
  __syncthreads();

  // Phase 2a: bias + scale + mask
  for (int idx = tid; idx < 16 * kT; idx += 256) {
    int r = idx >> 9, c = idx & (kT - 1);
    float val = S[r][c];
    val += dense[((size_t)b * kT + (q0 + r)) * kT + c] *
           ksum[((size_t)b * kT + c) * kH + hh];
    val *= 0.125f;  // rsqrt(A=64)
    float m = masks[((size_t)b * kT + (q0 + r)) * kT + c];
    val = val * m + (1.0f - ceilf(m)) * kNegInf;
    S[r][c] = val;
  }
  __syncthreads();

  // Phase 2b: softmax (16 threads per row, width-16 shuffle reduce)
  {
    int r = tid >> 4, j = tid & 15;
    float mx = kNegInf;
    for (int c = j; c < kT; c += 16) mx = fmaxf(mx, S[r][c]);
#pragma unroll
    for (int o = 1; o < 16; o <<= 1) mx = fmaxf(mx, __shfl_xor(mx, o, 16));
    float sum = 0.0f;
    for (int c = j; c < kT; c += 16) { float e = __expf(S[r][c] - mx); sum += e; S[r][c] = e; }
#pragma unroll
    for (int o = 1; o < 16; o <<= 1) sum += __shfl_xor(sum, o, 16);
    float inv = 1.0f / sum;
    for (int c = j; c < kT; c += 16) P[r][c] = (bf16)(S[r][c] * inv);
  }
  __syncthreads();

  // Phase 3: ctx[16,64] = P[16,512] @ V[512,64]; waves 0..3 one 16-col tile each
  if (wave < 4) {
    v8f acc = v8f{0};
    const bf16* Vt = vT + ((size_t)(b * kH + hh) * kA) * kT;  // [A][T], K contiguous
    for (int k0 = 0; k0 < kT; k0 += 32) {
      v16bf af  = frag_a((const bf16*)&P[0][0], 520, 0, k0, lane);  // LDS reads
      v16bf bfj = frag_bt(Vt, kT, k0, wave * 16, lane);
      acc = wmma_bf16(af, bfj, acc);
    }
    int col = hh * kA + wave * 16 + (lane & 15);
    int rb = (lane >> 4) << 3;
#pragma unroll
    for (int i = 0; i < 8; ++i)
      ctx[((size_t)b * kT + q0 + rb + i) * kHA + col] = (bf16)acc[i];
  }
}

// ---------------- host orchestration ----------------------------------------
extern "C" void kernel_launch(void* const* d_in, const int* in_sizes, int n_in,
                              void* d_out, int out_size, void* d_ws, size_t ws_size,
                              hipStream_t stream) {
  const int*   tokens = (const int*)d_in[0];
  const float* masks  = (const float*)d_in[1];
  const int*   ab     = (const int*)d_in[2];
  const float* emb    = (const float*)d_in[3];
  const float* Wq     = (const float*)d_in[4];
  const float* Wk     = (const float*)d_in[5];
  const float* Wv     = (const float*)d_in[6];
  const float* Wo     = (const float*)d_in[7];
  const float* be     = (const float*)d_in[8];
  const float* bs     = (const float*)d_in[9];
  const float* ln_g   = (const float*)d_in[10];
  const float* ln_b   = (const float*)d_in[11];
  const float* ff1_w  = (const float*)d_in[12];
  const float* ff1_b  = (const float*)d_in[13];
  const float* ff2_w  = (const float*)d_in[14];
  const float* ff2_b  = (const float*)d_in[15];
  const float* lng_o  = (const float*)d_in[16];
  const float* lnb_o  = (const float*)d_in[17];
  float* out = (float*)d_out;
  const int E = in_sizes[2] / 4;

  // workspace carve-up (256B aligned)
  char* p = (char*)d_ws;
  auto alloc = [&](size_t bytes) -> void* {
    void* r = (void*)p;
    p += (bytes + 255) & ~(size_t)255;
    return r;
  };
  const size_t wAttn = (size_t)kL * kD * kHA;        // per weight, elems
  const size_t wFF   = (size_t)kL * kD * kF;
  bf16* wqT  = (bf16*)alloc(wAttn * 2);
  bf16* wkT  = (bf16*)alloc(wAttn * 2);
  bf16* wvT  = (bf16*)alloc(wAttn * 2);
  bf16* woT  = (bf16*)alloc(wAttn * 2);
  bf16* w1T  = (bf16*)alloc(wFF * 2);
  bf16* w2T  = (bf16*)alloc(wFF * 2);
  float* x   = (float*)alloc((size_t)kRows * kD * 4);
  bf16* h_b  = (bf16*)alloc((size_t)kRows * kD * 2);
  bf16* q_b  = (bf16*)alloc((size_t)kRows * kHA * 2);
  bf16* k_b  = (bf16*)alloc((size_t)kRows * kHA * 2);
  bf16* v_b  = (bf16*)alloc((size_t)kRows * kHA * 2);
  bf16* vT_b = (bf16*)alloc((size_t)kRows * kHA * 2);
  bf16* ctx_b= (bf16*)alloc((size_t)kRows * kHA * 2);
  bf16* ff_b = (bf16*)alloc((size_t)kRows * kF * 2);
  float* ksumb = (float*)alloc((size_t)kB * kT * kH * 4);
  float* dense = (float*)alloc((size_t)kB * kT * kT * 4);

  auto grid1 = [](size_t n) { return dim3((unsigned)((n + 255) / 256)); };

  // one-time: weight convert+transpose (per layer; W[K,N] -> Wt[N,K] bf16)
  for (int l = 0; l < kL; ++l) {
    size_t oA = (size_t)l * kD * kHA;
    size_t oF = (size_t)l * kD * kF;
    convT_kernel<<<grid1(kD * kHA), 256, 0, stream>>>(Wq + oA, wqT + oA, kD, kHA);
    convT_kernel<<<grid1(kD * kHA), 256, 0, stream>>>(Wk + oA, wkT + oA, kD, kHA);
    convT_kernel<<<grid1(kD * kHA), 256, 0, stream>>>(Wv + oA, wvT + oA, kD, kHA);
    convT_kernel<<<grid1(kD * kHA), 256, 0, stream>>>(Wo + oA, woT + oA, kHA, kD);
    convT_kernel<<<grid1(kD * kF), 256, 0, stream>>>(ff1_w + oF, w1T + oF, kD, kF);
    convT_kernel<<<grid1(kD * kF), 256, 0, stream>>>(ff2_w + oF, w2T + oF, kF, kD);
  }

  // embedding + positional encoding
  embed_kernel<<<grid1((size_t)kRows * kD), 256, 0, stream>>>(tokens, emb, x);

  for (int l = 0; l < kL; ++l) {
    const float* g0 = ln_g + ((size_t)l * 2 + 0) * kD;
    const float* b0 = ln_b + ((size_t)l * 2 + 0) * kD;
    const float* g1 = ln_g + ((size_t)l * 2 + 1) * kD;
    const float* b1 = ln_b + ((size_t)l * 2 + 1) * kD;
    size_t oA = (size_t)l * kD * kHA;
    size_t oF = (size_t)l * kD * kF;

    // h = LN(x) -> bf16
    ln_kernel<true><<<dim3(kRows / 8), 256, 0, stream>>>(x, g0, b0, nullptr, h_b);

    // q,k,v = h @ W
    dim3 gq(kHA / 256, kRows / 64);
    gemm_kernel<0><<<gq, 256, 0, stream>>>(h_b, wqT + oA, nullptr, nullptr, q_b, kRows, kHA, kD);
    gemm_kernel<0><<<gq, 256, 0, stream>>>(h_b, wkT + oA, nullptr, nullptr, k_b, kRows, kHA, kD);
    gemm_kernel<0><<<gq, 256, 0, stream>>>(h_b, wvT + oA, nullptr, nullptr, v_b, kRows, kHA, kD);

    ksum_kernel<<<grid1((size_t)kB * kT * kH), 256, 0, stream>>>(k_b, ksumb);
    vtrans_kernel<<<grid1((size_t)kRows * kHA), 256, 0, stream>>>(v_b, vT_b);

    // edge bias -> dense[B,T,T]
    fill_zero_kernel<<<grid1((size_t)kB * kT * kT), 256, 0, stream>>>(dense, kB * kT * kT);
    bias_scatter_kernel<<<grid1((size_t)E), 256, 0, stream>>>(
        ab, be + (size_t)l * kBiasDim * kA, bs + (size_t)l * kA, dense, E);

    // fused attention
    attn_kernel<<<dim3(kB * kH * (kT / 16)), 256, 0, stream>>>(
        q_b, k_b, vT_b, dense, ksumb, masks, ctx_b);

    // x += ctx @ Wo
    gemm_kernel<1><<<dim3(kD / 256, kRows / 64), 256, 0, stream>>>(
        ctx_b, woT + oA, nullptr, x, nullptr, kRows, kD, kHA);

    // h = LN(x); ff = relu(h @ W1 + b1); x += ff @ W2 + b2
    ln_kernel<true><<<dim3(kRows / 8), 256, 0, stream>>>(x, g1, b1, nullptr, h_b);
    gemm_kernel<2><<<dim3(kF / 256, kRows / 64), 256, 0, stream>>>(
        h_b, w1T + oF, ff1_b + (size_t)l * kF, nullptr, ff_b, kRows, kF, kD);
    gemm_kernel<1><<<dim3(kD / 256, kRows / 64), 256, 0, stream>>>(
        ff_b, w2T + oF, ff2_b + (size_t)l * kD, x, nullptr, kRows, kD, kF);
  }

  // final LN -> f32 output
  ln_kernel<false><<<dim3(kRows / 8), 256, 0, stream>>>(x, lng_o, lnb_o, out, nullptr);
}
